// MultiheadAttention_8263517078082
// MI455X (gfx1250) — matrix operand order repaired and checked by reading, not compiled, not run on hardware
//
#include <hip/hip_runtime.h>
#include <hip/hip_bf16.h>
#include <stdint.h>

// ---------------- problem constants ----------------
#define BATCH 2
#define SEQ   2048
#define DMODEL 1024
#define NHEAD 16
#define HSZ   64
#define M1    (BATCH * SEQ)          // 4096 rows for projections
#define N_QKV (3 * DMODEL)           // 3072
#define OUT_ELEMS ((size_t)BATCH * SEQ * DMODEL)               // 4,194,304

// ---------------- CDNA5 feature detection (device pass only) ----------------
#if defined(__gfx1250__) && __has_builtin(__builtin_amdgcn_global_load_async_to_lds_b128)
#define USE_ASYNC_LDS 1
#endif
#if defined(__gfx1250__) && __has_builtin(__builtin_amdgcn_tensor_load_to_lds)
#define USE_TDM 1
#endif

// ---------------- CDNA5 WMMA types ----------------
typedef __attribute__((ext_vector_type(16))) __bf16 v16bf;
typedef __attribute__((ext_vector_type(8)))  __bf16 v8bf;
typedef __attribute__((ext_vector_type(8)))  float  v8f;
typedef __attribute__((ext_vector_type(4)))  unsigned int v4u;
typedef __attribute__((ext_vector_type(8)))  int v8i;
typedef __attribute__((ext_vector_type(4)))  int v4i;
// gcc-style vector, exactly matching the async-LDS builtin's pointee type
typedef int v4i_vs __attribute__((vector_size(16)));

__device__ __forceinline__ v8f wmma_bf16(v16bf a, v16bf b, v8f c) {
  // 8 args: (neg_a, A, neg_b, B, c_mod, C, reuse_a, reuse_b)
  return __builtin_amdgcn_wmma_f32_16x16x32_bf16(false, a, false, b, (short)0, c,
                                                 false, false);
}

__device__ __forceinline__ unsigned short f32_to_bf16(float f) {
  unsigned u = __float_as_uint(f);
  u = u + 0x7FFFu + ((u >> 16) & 1u);   // round-to-nearest-even
  return (unsigned short)(u >> 16);
}

#ifdef USE_ASYNC_LDS
// one 16B lane-copy global->LDS, tracked by ASYNCcnt
__device__ __forceinline__ void async_cp16(void* lds, const void* g) {
  __builtin_amdgcn_global_load_async_to_lds_b128(
      (__attribute__((address_space(1))) v4i_vs*)g,
      (__attribute__((address_space(3))) v4i_vs*)lds, 0, 0);
}
template <int N> __device__ __forceinline__ void wait_async() {
#if __has_builtin(__builtin_amdgcn_s_wait_asynccnt)
  __builtin_amdgcn_s_wait_asynccnt(N);
#else
  asm volatile("s_wait_asynccnt %0" ::"i"(N) : "memory");
#endif
}
#endif

#ifdef USE_TDM
__device__ __forceinline__ unsigned lds_offset_of(const void* p) {
  return (unsigned)(unsigned long long)(__attribute__((address_space(3))) const char*)p;
}
// 1-row TDM tile: copy `n2` bf16 elements (contiguous) from global to LDS.
__device__ __forceinline__ void tdm_copy_1d(void* lds, const void* gsrc, unsigned n2) {
  unsigned long long ga = (unsigned long long)gsrc;
  unsigned lo = lds_offset_of(lds);
  // D# group0: count=1 | lds_addr | global_addr[56:0] | type=2 (bits 127:126)
  v4u g0 = { 1u, lo, (unsigned)ga,
             (unsigned)((ga >> 32) & 0x01FFFFFFu) | 0x80000000u };
  // D# group1: data_size=2B (enc 1); tensor_dim0=n2; tensor_dim1=1;
  //            tile_dim0=n2; tile_dim1=1; stride0=n2
  v8i g1 = { (int)(1u << 16),                    // workgroup_mask=0, data_size=1(2B)
             (int)((n2 & 0xFFFFu) << 16),        // tensor_dim0[15:0] @ bits63:48
             (int)((n2 >> 16) | (1u << 16)),     // tensor_dim0[31:16] | tensor_dim1=1
             (int)((n2 & 0xFFFFu) << 16),        // tile_dim0 @ bits127:112
             (int)1,                             // tile_dim1=1, tile_dim2=0
             (int)n2,                            // tensor_dim0_stride[31:0]
             0, 0 };
  v4i z4 = { 0, 0, 0, 0 };
  v8i z8 = { 0, 0, 0, 0, 0, 0, 0, 0 };
  // 6-arg toolchain form: (g0, g1, g2, g3, g4, cpol)
  __builtin_amdgcn_tensor_load_to_lds(g0, g1, z4, z4, z8, 0);
}
#endif

// ---- A fragment (16x32 bf16, M x K) from LDS, per ISA 16-bit A layout:
//  lane m = lane&15 ; kbase = (lane&16)?8:0
//  halves 0..7  -> K = kbase   .. kbase+7   (contiguous 16B)
//  halves 8..15 -> K = kbase+16.. kbase+23  (contiguous 16B)
__device__ __forceinline__ v16bf load_frag_a(const unsigned short* As, int lda, int lane) {
  int m  = lane & 15;
  int kb = (lane & 16) ? 8 : 0;
  v8bf lo = *(const v8bf*)(As + (size_t)m * lda + kb);
  v8bf hi = *(const v8bf*)(As + (size_t)m * lda + kb + 16);
  v16bf a;
#pragma unroll
  for (int i = 0; i < 8; ++i) { a[i] = lo[i]; a[i + 8] = hi[i]; }
  return a;
}

// ---- B fragment (32x16 bf16, K x N) with B staged in LDS as [n][k]:
//  lane n = lane&15 ; kbase = (lane&16)?8:0 ; halves 0..7 -> K=kbase..kbase+7, 8..15 -> K=kbase+16..kbase+23
__device__ __forceinline__ v16bf load_frag_b(const unsigned short* Bs, int ldb, int lane) {
  int n    = lane & 15;
  int kb   = (lane & 16) ? 8 : 0;
  v8bf lo = *(const v8bf*)(Bs + (size_t)n * ldb + kb);
  v8bf hi = *(const v8bf*)(Bs + (size_t)n * ldb + kb + 16);
  v16bf b;
#pragma unroll
  for (int i = 0; i < 8; ++i) { b[i] = lo[i]; b[i + 8] = hi[i]; }
  return b;
}

// ================= fp32 -> bf16 convert =================
__global__ void __launch_bounds__(256) cvt_bf16_kernel(const float* __restrict__ in,
                                                       unsigned short* __restrict__ out,
                                                       int n) {
  int i = blockIdx.x * 256 + threadIdx.x;
  if (i < n) out[i] = f32_to_bf16(in[i]);
}

// ================= NT GEMM: C[m][n] = sum_k A[m][k]*B[n][k] + bias[n] =================
// mode 0: write fp32 C (ldc = N)      mode 1: QKV epilogue -> scatter bf16 Q(*1/8)/K/V
__global__ void __launch_bounds__(256) gemm_nt_bf16(const unsigned short* __restrict__ A,
                                                    const unsigned short* __restrict__ Bm,
                                                    const float* __restrict__ bias,
                                                    int K, float* __restrict__ Cout, int ldc,
                                                    unsigned short* __restrict__ Qb,
                                                    unsigned short* __restrict__ Kb,
                                                    unsigned short* __restrict__ Vb,
                                                    int mode) {
  __shared__ alignas(32) unsigned short As[2][128][32];
  __shared__ alignas(32) unsigned short Bs[2][128][32];

  const int tid  = threadIdx.x;
  const int lane = tid & 31;
  const int wave = tid >> 5;        // 8 waves
  const int wm   = wave >> 2;       // 0..1 -> 64 rows each
  const int wn   = wave & 3;        // 0..3 -> 32 cols each
  const int m0   = blockIdx.y * 128;
  const int n0   = blockIdx.x * 128;

  v8f acc[4][2];
#pragma unroll
  for (int i = 0; i < 4; ++i)
#pragma unroll
    for (int j = 0; j < 2; ++j) acc[i][j] = (v8f){};

#ifdef USE_ASYNC_LDS
  // ---------- double-buffered pipeline via GLOBAL_LOAD_ASYNC_TO_LDS_B128 ----------
  // each wave issues 4 async copy instructions per tile (2 for A, 2 for B)
  {
#pragma unroll
    for (int c = tid; c < 512; c += 256) {
      int r = c >> 2, col = (c & 3) << 3;
      async_cp16(&As[0][r][col], A  + (size_t)(m0 + r) * K + col);
      async_cp16(&Bs[0][r][col], Bm + (size_t)(n0 + r) * K + col);
    }
  }
  int p = 0;
  for (int k0 = 0; k0 < K; k0 += 32, p ^= 1) {
    const bool has_next = (k0 + 32) < K;
    if (has_next) {
#pragma unroll
      for (int c = tid; c < 512; c += 256) {
        int r = c >> 2, col = (c & 3) << 3;
        async_cp16(&As[p ^ 1][r][col], A  + (size_t)(m0 + r) * K + k0 + 32 + col);
        async_cp16(&Bs[p ^ 1][r][col], Bm + (size_t)(n0 + r) * K + k0 + 32 + col);
      }
      wait_async<4>();   // allow the 4 just-issued; current buffer's copies complete
    } else {
      wait_async<0>();
    }
    __syncthreads();

    v16bf bfr[2];
#pragma unroll
    for (int ni = 0; ni < 2; ++ni)
      bfr[ni] = load_frag_b(&Bs[p][wn * 32 + ni * 16][0], 32, lane);
#pragma unroll
    for (int mi = 0; mi < 4; ++mi) {
      v16bf af = load_frag_a(&As[p][wm * 64 + mi * 16][0], 32, lane);
#pragma unroll
      for (int ni = 0; ni < 2; ++ni)
        acc[mi][ni] = wmma_bf16(af, bfr[ni], acc[mi][ni]);
    }
    __syncthreads();
  }
#else
  // ---------- fallback: synchronous staging ----------
  for (int k0 = 0; k0 < K; k0 += 32) {
#pragma unroll
    for (int c = tid; c < 512; c += 256) {
      int r = c >> 2, col = (c & 3) << 3;
      *(v8bf*)&As[0][r][col] = *(const v8bf*)(A  + (size_t)(m0 + r) * K + k0 + col);
      *(v8bf*)&Bs[0][r][col] = *(const v8bf*)(Bm + (size_t)(n0 + r) * K + k0 + col);
    }
    __syncthreads();
    v16bf bfr[2];
#pragma unroll
    for (int ni = 0; ni < 2; ++ni)
      bfr[ni] = load_frag_b(&Bs[0][wn * 32 + ni * 16][0], 32, lane);
#pragma unroll
    for (int mi = 0; mi < 4; ++mi) {
      v16bf af = load_frag_a(&As[0][wm * 64 + mi * 16][0], 32, lane);
#pragma unroll
      for (int ni = 0; ni < 2; ++ni)
        acc[mi][ni] = wmma_bf16(af, bfr[ni], acc[mi][ni]);
    }
    __syncthreads();
  }
#endif

  // epilogue: C/D layout -> n = lane&15, rows r + 8*(lane>=16)
  const int nloc = lane & 15;
  const int mhi  = (lane >> 4) << 3;
#pragma unroll
  for (int mi = 0; mi < 4; ++mi) {
#pragma unroll
    for (int ni = 0; ni < 2; ++ni) {
      int ng = n0 + wn * 32 + ni * 16 + nloc;
      float bv = bias[ng];
#pragma unroll
      for (int r = 0; r < 8; ++r) {
        int mg = m0 + wm * 64 + mi * 16 + mhi + r;
        float v = acc[mi][ni][r] + bv;
        if (mode == 0) {
          Cout[(size_t)mg * ldc + ng] = v;
        } else {
          int seg = ng >> 10, f = ng & 1023;
          int h = f >> 6, dd = f & 63;
          int bb = mg >> 11, s = mg & 2047;
          size_t idx = (((size_t)bb * NHEAD + h) * SEQ + s) * HSZ + dd;
          if (seg == 0)      Qb[idx] = f32_to_bf16(v * 0.125f);  // 1/sqrt(64)
          else if (seg == 1) Kb[idx] = f32_to_bf16(v);
          else               Vb[idx] = f32_to_bf16(v);
        }
      }
    }
  }
}

// ================= causal scores: attn = (Q/sqrt(d)) @ K^T, lower-tri tiles only ==========
__global__ void __launch_bounds__(256) attn_scores_kernel(const unsigned short* __restrict__ Qb,
                                                          const unsigned short* __restrict__ Kb,
                                                          float* __restrict__ attn) {
  __shared__ alignas(32) unsigned short Qs[128][64];
  __shared__ alignas(32) unsigned short Ks[128][64];

  const int bh  = blockIdx.y;
  int idx = blockIdx.x;                 // 0..135 lower-triangular 128x128 tiles
  int qt = 0, cum = 0;
  while (idx >= cum + qt + 1) { cum += qt + 1; ++qt; }
  const int kt = idx - cum;

  const int tid  = threadIdx.x;
  const int lane = tid & 31;
  const int wave = tid >> 5;
  const int wm   = wave >> 2;  // 0..1
  const int wn   = wave & 3;   // 0..3

  const unsigned short* Qh = Qb + (size_t)bh * SEQ * HSZ;
  const unsigned short* Kh = Kb + (size_t)bh * SEQ * HSZ;

#ifdef USE_TDM
  // each 128x64 bf16 head tile is one contiguous 16KB block -> one TDM op each
  if (wave == 0) {
    tdm_copy_1d(&Qs[0][0], Qh + (size_t)(qt * 128) * HSZ, 128 * HSZ);
    tdm_copy_1d(&Ks[0][0], Kh + (size_t)(kt * 128) * HSZ, 128 * HSZ);
    __builtin_amdgcn_s_wait_tensorcnt(0);
  }
#else
#pragma unroll
  for (int c = tid; c < 1024; c += 256) {
    int r = c >> 3, col = (c & 7) << 3;
    *(v8bf*)&Qs[r][col] = *(const v8bf*)(Qh + (size_t)(qt * 128 + r) * HSZ + col);
    *(v8bf*)&Ks[r][col] = *(const v8bf*)(Kh + (size_t)(kt * 128 + r) * HSZ + col);
  }
#endif
  __syncthreads();

  const int nloc = lane & 15;
  const int mhi  = (lane >> 4) << 3;
#pragma unroll
  for (int mi = 0; mi < 4; ++mi) {
#pragma unroll
    for (int ni = 0; ni < 2; ++ni) {
      v8f c = (v8f){};
#pragma unroll
      for (int ks = 0; ks < 64; ks += 32) {
        v16bf a = load_frag_a(&Qs[wm * 64 + mi * 16][ks], 64, lane);
        v16bf b = load_frag_b(&Ks[wn * 32 + ni * 16][ks], 64, lane);
        c = wmma_bf16(a, b, c);
      }
      int kk = kt * 128 + wn * 32 + ni * 16 + nloc;
#pragma unroll
      for (int r = 0; r < 8; ++r) {
        int q = qt * 128 + wm * 64 + mi * 16 + mhi + r;
        attn[((size_t)bh * SEQ + q) * SEQ + kk] = c[r];
      }
    }
  }
}

// ================= causal softmax in place (writes exact zeros above diagonal) ============
__global__ void __launch_bounds__(256) softmax_causal_kernel(float* __restrict__ attn) {
  const size_t rid = blockIdx.x;                // 0 .. B*H*S-1
  const int q = (int)(rid & (SEQ - 1));
  float* row = attn + rid * SEQ;
  const int L = q + 1;
  const int t = threadIdx.x, lane = t & 31, wave = t >> 5;

  __shared__ float sm[8];
  float vals[8];
  float vmax = -__builtin_inff();
#pragma unroll
  for (int i = 0; i < 8; ++i) {
    int k = t + i * 256;
    float v = (k < L) ? row[k] : -__builtin_inff();
    vals[i] = v;
    vmax = fmaxf(vmax, v);
  }
#pragma unroll
  for (int off = 16; off > 0; off >>= 1) vmax = fmaxf(vmax, __shfl_xor(vmax, off, 32));
  if (lane == 0) sm[wave] = vmax;
  __syncthreads();
  vmax = sm[0];
#pragma unroll
  for (int w = 1; w < 8; ++w) vmax = fmaxf(vmax, sm[w]);
  __syncthreads();

  float vsum = 0.f;
#pragma unroll
  for (int i = 0; i < 8; ++i) {
    float e = __expf(vals[i] - vmax);   // exp(-inf) == 0 above the diagonal
    vals[i] = e;
    vsum += e;
  }
#pragma unroll
  for (int off = 16; off > 0; off >>= 1) vsum += __shfl_xor(vsum, off, 32);
  if (lane == 0) sm[wave] = vsum;
  __syncthreads();
  vsum = 0.f;
#pragma unroll
  for (int w = 0; w < 8; ++w) vsum += sm[w];
  const float inv = 1.0f / vsum;
#pragma unroll
  for (int i = 0; i < 8; ++i) row[t + i * 256] = vals[i] * inv;
}

// ================= attn @ V  (K-loop truncated by causality) =================
__global__ void __launch_bounds__(256) attn_av_kernel(const float* __restrict__ attn,
                                                      const unsigned short* __restrict__ Vb,
                                                      unsigned short* __restrict__ AOb) {
  __shared__ alignas(32) unsigned short Ps[128][32];   // probs tile (bf16)
  __shared__ alignas(32) unsigned short Vst[64][32];   // V tile transposed [d][k]

  const int bh = blockIdx.y;
  const int qt = blockIdx.x;            // 0..15
  const int tid = threadIdx.x, lane = tid & 31, wave = tid >> 5;

  v8f acc[4];
#pragma unroll
  for (int i = 0; i < 4; ++i) acc[i] = (v8f){};

  const int kmax = (qt + 1) * 128;      // probs are exactly 0 beyond the diagonal
  const float* Prow = attn + ((size_t)bh * SEQ + qt * 128) * SEQ;
  const unsigned short* Vh = Vb + (size_t)bh * SEQ * HSZ;

  for (int kk = 0; kk < kmax; kk += 32) {
    {   // fp32 probs -> bf16 LDS (128x32), 16 elems/thread
      int r = tid >> 1, c0 = (tid & 1) << 4;
      const float* src = Prow + (size_t)r * SEQ + kk + c0;
      if (kk + 32 < kmax) __builtin_prefetch(src + 32, 0, 0);  // -> global_prefetch
#pragma unroll
      for (int j = 0; j < 16; ++j) Ps[r][c0 + j] = f32_to_bf16(src[j]);
    }
    {   // V (32x64) -> transposed LDS (64x32), 8 elems/thread
      int kr = tid >> 3, c0 = (tid & 7) << 3;
      const unsigned short* vs = Vh + (size_t)(kk + kr) * HSZ + c0;
#pragma unroll
      for (int j = 0; j < 8; ++j) Vst[c0 + j][kr] = vs[j];
    }
    __syncthreads();

    v16bf a = load_frag_a(&Ps[wave * 16][0], 32, lane);
#pragma unroll
    for (int ni = 0; ni < 4; ++ni) {
      v16bf b = load_frag_b(&Vst[ni * 16][0], 32, lane);
      acc[ni] = wmma_bf16(a, b, acc[ni]);
    }
    __syncthreads();
  }

  const int bb = bh >> 4, h = bh & 15;
  const int nloc = lane & 15, mhi = (lane >> 4) << 3;
#pragma unroll
  for (int ni = 0; ni < 4; ++ni) {
#pragma unroll
    for (int r = 0; r < 8; ++r) {
      int q  = qt * 128 + wave * 16 + mhi + r;
      int dd = ni * 16 + nloc;
      AOb[((size_t)bb * SEQ + q) * DMODEL + h * HSZ + dd] = f32_to_bf16(acc[ni][r]);
    }
  }
}

// ================= launch =================
extern "C" void kernel_launch(void* const* d_in, const int* in_sizes, int n_in,
                              void* d_out, int out_size, void* d_ws, size_t ws_size,
                              hipStream_t stream) {
  const float* x     = (const float*)d_in[0];
  const float* w_in  = (const float*)d_in[1];
  const float* b_in  = (const float*)d_in[2];
  const float* w_out = (const float*)d_in[3];
  const float* b_out = (const float*)d_in[4];

  float* out  = (float*)d_out;                 // (B,S,D)
  float* attn = (float*)d_out + OUT_ELEMS;     // (B,H,S,S)

  // workspace carve-out (bf16 buffers), ~48 MB total
  char* ws = (char*)d_ws;
  unsigned short* Xb   = (unsigned short*)(ws);                     // 4096x1024
  unsigned short* Wqb  = (unsigned short*)(ws + (8u << 20));        // 3072x1024
  unsigned short* Wob  = (unsigned short*)(ws + (14u << 20));       // 1024x1024
  unsigned short* Qb   = (unsigned short*)(ws + (16u << 20));       // (B,H,S,64)
  unsigned short* Kb   = (unsigned short*)(ws + (24u << 20));
  unsigned short* Vb   = (unsigned short*)(ws + (32u << 20));
  unsigned short* AOb  = (unsigned short*)(ws + (40u << 20));       // (B,S,D)

  const int BH = BATCH * NHEAD;

  // 1) converts
  cvt_bf16_kernel<<<(M1 * DMODEL + 255) / 256, 256, 0, stream>>>(x, Xb, M1 * DMODEL);
  cvt_bf16_kernel<<<(N_QKV * DMODEL + 255) / 256, 256, 0, stream>>>(w_in, Wqb, N_QKV * DMODEL);
  cvt_bf16_kernel<<<(DMODEL * DMODEL + 255) / 256, 256, 0, stream>>>(w_out, Wob, DMODEL * DMODEL);

  // 2) QKV projection (mode 1 epilogue: scale Q, scatter to head-major bf16)
  gemm_nt_bf16<<<dim3(N_QKV / 128, M1 / 128), 256, 0, stream>>>(
      Xb, Wqb, b_in, DMODEL, nullptr, 0, Qb, Kb, Vb, 1);

  // 3) causal score tiles -> attn (fp32)
  attn_scores_kernel<<<dim3(136, BH), 256, 0, stream>>>(Qb, Kb, attn);

  // 4) causal softmax in place
  softmax_causal_kernel<<<(unsigned)(BH * SEQ), 256, 0, stream>>>(attn);

  // 5) attn @ V -> AOb (bf16, heads concatenated)
  attn_av_kernel<<<dim3(SEQ / 128, BH), 256, 0, stream>>>(attn, Vb, AOb);

  // 6) output projection (mode 0: fp32 C)
  gemm_nt_bf16<<<dim3(DMODEL / 128, M1 / 128), 256, 0, stream>>>(
      AOb, Wob, b_out, DMODEL, out, DMODEL, nullptr, nullptr, nullptr, 0);
}
